// Event_Critic_Net_51539607595
// MI455X (gfx1250) — matrix operand order repaired and checked by reading, not compiled, not run on hardware
//
#include <hip/hip_runtime.h>
#include <math.h>

// ---------------- problem constants (from reference) ----------------
constexpr int N   = 131072;     // nodes per branch
constexpr int G   = 4096;       // graphs
constexpr int NPG = 32;         // nodes per graph
constexpr int E   = 1048576;    // edges per branch
constexpr int S   = 64;         // state size
constexpr int H   = 128;        // hidden
constexpr int CAP = 131072;     // compact edge capacity (expected ~E/32 = 32768)
constexpr float NEG_SLOPE = 0.2f;

typedef __attribute__((ext_vector_type(16))) __bf16 v16bf;
typedef __attribute__((ext_vector_type(8)))  float  v8f;

// ---------------- workspace layout (bytes) ----------------
// wvec: 4 x 64 floats (W@att_src, W@att_dst for up and down)
constexpr size_t WVEC_SZ = 1024;
constexpr size_t BR_ASRC = 0;                         // N f32
constexpr size_t BR_ADST = BR_ASRC + (size_t)N * 4;   // N f32
constexpr size_t BR_MENC = BR_ADST + (size_t)N * 4;   // G u32 (ordered-uint encoded max)
constexpr size_t BR_DEN  = BR_MENC + (size_t)G * 4;   // G f32 softmax denominators
constexpr size_t BR_CNT  = BR_DEN  + (size_t)G * 4;   // 1 u32 counter (padded)
constexpr size_t BR_CSRC = BR_CNT  + 256;             // CAP i32 compact src
constexpr size_t BR_CG   = BR_CSRC + (size_t)CAP * 4; // CAP i32 compact graph id
constexpr size_t BR_CP   = BR_CG   + (size_t)CAP * 4; // CAP f32 logit -> p
constexpr size_t BR_XAGG = BR_CP   + (size_t)CAP * 4; // G*S f32 aggregated features
constexpr size_t BR_SIG  = BR_XAGG + (size_t)G * S * 4; // G*H f32 sigmoid branch output
constexpr size_t BR_SZ   = BR_SIG  + (size_t)G * H * 4;
constexpr size_t WS_TOTAL = WVEC_SZ + 2 * BR_SZ;      // ~11.6 MB

// ---------------- helpers ----------------
__device__ __forceinline__ unsigned enc_f(float f) {  // order-preserving float->uint
  unsigned u = __float_as_uint(f);
  return (u & 0x80000000u) ? ~u : (u | 0x80000000u);
}
__device__ __forceinline__ float dec_f(unsigned u) {
  u = (u & 0x80000000u) ? (u & 0x7fffffffu) : ~u;
  return __uint_as_float(u);
}

// ---------------- kernels ----------------
__global__ void k_zero(float* p, size_t n) {
  size_t i = (size_t)blockIdx.x * blockDim.x + threadIdx.x;
  size_t stride = (size_t)gridDim.x * blockDim.x;
  for (; i < n; i += stride) p[i] = 0.f;
}

// wvec[0]=up W@att_src, [1]=up W@att_dst, [2]=down W@att_src, [3]=down W@att_dst
__global__ void k_prep(const float* __restrict__ upW, const float* __restrict__ upAs,
                       const float* __restrict__ upAd, const float* __restrict__ dnW,
                       const float* __restrict__ dnAs, const float* __restrict__ dnAd,
                       float* __restrict__ wvec) {
  int k = threadIdx.x;
  if (k >= S) return;
  float s0 = 0.f, s1 = 0.f, s2 = 0.f, s3 = 0.f;
  for (int j = 0; j < H; ++j) {
    float wu = upW[k * H + j], wd = dnW[k * H + j];
    s0 += wu * upAs[j]; s1 += wu * upAd[j];
    s2 += wd * dnAs[j]; s3 += wd * dnAd[j];
  }
  wvec[0 * S + k] = s0; wvec[1 * S + k] = s1;
  wvec[2 * S + k] = s2; wvec[3 * S + k] = s3;
}

// a_src[n] = x[n]·(W@att_src), a_dst[n] = x[n]·(W@att_dst): 16 lanes per node
__global__ void k_av(const float* __restrict__ x, const float* __restrict__ wsrc,
                     const float* __restrict__ wdst, float* __restrict__ a_src,
                     float* __restrict__ a_dst) {
  int t = blockIdx.x * blockDim.x + threadIdx.x;
  int node = t >> 4;
  int l = t & 15;
  if (node >= N) return;
  float4 xv = *(const float4*)(x + (size_t)node * S + l * 4);
  float4 ws = *(const float4*)(wsrc + l * 4);
  float4 wd = *(const float4*)(wdst + l * 4);
  float ps = xv.x * ws.x + xv.y * ws.y + xv.z * ws.z + xv.w * ws.w;
  float pd = xv.x * wd.x + xv.y * wd.y + xv.z * wd.z + xv.w * wd.w;
  for (int o = 8; o > 0; o >>= 1) {
    ps += __shfl_xor(ps, o, 16);
    pd += __shfl_xor(pd, o, 16);
  }
  if (l == 0) { a_src[node] = ps; a_dst[node] = pd; }
}

// Sweep all edges; only dst%32==31 feeds the final output. Compact those edges
// and run the segment max with ordered-uint atomics on G-sized (L2-resident) state.
__global__ void k_edge_filter(const int* __restrict__ ei,
                              const float* __restrict__ a_src,
                              const float* __restrict__ a_dst,
                              unsigned* __restrict__ m_enc,
                              unsigned* __restrict__ counter,
                              int* __restrict__ csrc, int* __restrict__ cg,
                              float* __restrict__ cp) {
  int e = blockIdx.x * blockDim.x + threadIdx.x;
  if (e >= E) return;
  __builtin_prefetch(ei + E + e + 32768, 0, 0);   // stream dst list ahead
  int dst = ei[E + e];
  if ((dst & (NPG - 1)) != NPG - 1) return;       // only last node of each graph
  int src = ei[e];
  float t = a_src[src] + a_dst[dst];
  float logit = t > 0.f ? t : NEG_SLOPE * t;
  int g = dst >> 5;
  atomicMax(m_enc + g, enc_f(logit));
  unsigned pos = atomicAdd(counter, 1u);
  if (pos < (unsigned)CAP) { csrc[pos] = src; cg[pos] = g; cp[pos] = logit; }
}

// p = exp(logit - m[g]); denom[g] += p   (compact edges only)
__global__ void k_edge_soft(const unsigned* __restrict__ counter,
                            const int* __restrict__ cg,
                            const unsigned* __restrict__ m_enc,
                            float* __restrict__ cp, float* __restrict__ denom) {
  unsigned c = *counter;
  unsigned n = c < (unsigned)CAP ? c : (unsigned)CAP;
  unsigned i = blockIdx.x * blockDim.x + threadIdx.x;
  if (i >= n) return;
  int g = cg[i];
  float p = __expf(cp[i] - dec_f(m_enc[g]));
  cp[i] = p;
  atomicAdd(denom + g, p);
}

// Aggregate in S=64 input space (linearity: (Σ alpha x[src]) @ W done later on WMMA)
__global__ void k_edge_agg(const unsigned* __restrict__ counter,
                           const int* __restrict__ csrc, const int* __restrict__ cg,
                           const float* __restrict__ cp, const float* __restrict__ denom,
                           const float* __restrict__ x, float* __restrict__ xagg) {
  unsigned c = *counter;
  unsigned n = c < (unsigned)CAP ? c : (unsigned)CAP;
  unsigned t = blockIdx.x * blockDim.x + threadIdx.x;
  unsigned i = t >> 4;
  int l = t & 15;
  if (i >= n) return;
  int g = cg[i];
  float alpha = cp[i] * __builtin_amdgcn_rcpf(denom[g] + 1e-16f);  // v_rcp_f32
  float4 xv = *(const float4*)(x + (size_t)csrc[i] * S + l * 4);
  float* dst = xagg + (size_t)g * S + l * 4;
  atomicAdd(dst + 0, alpha * xv.x);
  atomicAdd(dst + 1, alpha * xv.y);
  atomicAdd(dst + 2, alpha * xv.z);
  atomicAdd(dst + 3, alpha * xv.w);
}

// sig[g][:] = sigmoid(xagg[g] @ W + bias) via v_wmma_f32_16x16x32_bf16.
// Block: 16 graph rows x 128 cols; 8 waves = 8 column tiles; K=64 = 2 wmma steps.
__global__ void __launch_bounds__(256) k_gat_out(const float* __restrict__ xagg,
                                                 const float* __restrict__ W,
                                                 const float* __restrict__ bias,
                                                 float* __restrict__ sig) {
  __shared__ alignas(16) __bf16 sWt[H * S];  // transposed W: [n][k]
  __shared__ alignas(16) __bf16 sA[16 * S];  // 16 rows of xagg
  int tid = threadIdx.x;
  for (int idx = tid; idx < S * H; idx += 256) {
    int k = idx >> 7, n = idx & (H - 1);
    sWt[n * S + k] = (__bf16)W[idx];
  }
  int g0 = blockIdx.x * 16;
  for (int idx = tid; idx < 16 * S; idx += 256) {
    int r = idx >> 6, k = idx & (S - 1);
    sA[idx] = (__bf16)xagg[(size_t)(g0 + r) * S + k];
  }
  __syncthreads();

  int lane = tid & 31;
  int wave = tid >> 5;
  int n0 = wave * 16;
  int m = lane & 15, hi = lane >> 4;

  // A 16x32 bf16 per ISA: lanes 0-15 -> K[0..7],[16..23]; lanes 16-31 -> K[8..15],[24..31]
  union { v16bf v; uint4 u[2]; } a0, a1, b0, b1;
  {
    const __bf16* p = sA + m * S;
    a0.u[0] = *(const uint4*)(p + hi * 8);
    a0.u[1] = *(const uint4*)(p + 16 + hi * 8);
    a1.u[0] = *(const uint4*)(p + 32 + hi * 8);
    a1.u[1] = *(const uint4*)(p + 48 + hi * 8);
  }
  // B 32x16 bf16: lane = column; lanes 0-15 hold K[0..7],[16..23], lanes 16-31 hold K[8..15],[24..31]
  {
    const __bf16* q = sWt + (size_t)(n0 + m) * S;
    b0.u[0] = *(const uint4*)(q + hi * 8);
    b0.u[1] = *(const uint4*)(q + 16 + hi * 8);
    b1.u[0] = *(const uint4*)(q + 32 + hi * 8);
    b1.u[1] = *(const uint4*)(q + 48 + hi * 8);
  }
  v8f acc = {};
  acc = __builtin_amdgcn_wmma_f32_16x16x32_bf16(false, a0.v, false, b0.v,
                                                (short)0, acc, false, false);
  acc = __builtin_amdgcn_wmma_f32_16x16x32_bf16(false, a1.v, false, b1.v,
                                                (short)0, acc, false, false);
  // C/D layout: VGPR r -> M=r (lanes 0-15), M=8+r (lanes 16-31); N = lane&15
  int n = n0 + m;
  float bn = bias[n];
#pragma unroll
  for (int r = 0; r < 8; ++r) {
    int row = hi * 8 + r;
    float v = acc[r] + bn;
    // fast sigmoid: v_exp_f32 + v_rcp_f32 (avoid IEEE div chain)
    sig[(size_t)(g0 + row) * H + n] = __builtin_amdgcn_rcpf(1.f + __expf(-v));
  }
}

// out[g] = sum_f up[g][f]*down[g][f]*mlpW[f] + mlpB   (one wave per graph)
__global__ void k_final(const float* __restrict__ sup, const float* __restrict__ sdn,
                        const float* __restrict__ mlpW, const float* __restrict__ mlpB,
                        float* __restrict__ out) {
  int t = blockIdx.x * blockDim.x + threadIdx.x;
  int g = t >> 5;
  int l = t & 31;
  if (g >= G) return;
  float4 u = *(const float4*)(sup + (size_t)g * H + l * 4);
  float4 d = *(const float4*)(sdn + (size_t)g * H + l * 4);
  float4 w = *(const float4*)(mlpW + l * 4);
  float s = u.x * d.x * w.x + u.y * d.y * w.y + u.z * d.z * w.z + u.w * d.w * w.w;
  for (int o = 16; o > 0; o >>= 1) s += __shfl_xor(s, o, 32);
  if (l == 0) out[g] = s + mlpB[0];
}

// ---------------- launcher ----------------
extern "C" void kernel_launch(void* const* d_in, const int* in_sizes, int n_in,
                              void* d_out, int out_size, void* d_ws, size_t ws_size,
                              hipStream_t stream) {
  (void)in_sizes; (void)n_in; (void)out_size; (void)ws_size;
  const float* up_x   = (const float*)d_in[0];
  const int*   up_ei  = (const int*)d_in[1];
  const float* dn_x   = (const float*)d_in[3];
  const int*   dn_ei  = (const int*)d_in[4];
  const float* up_W   = (const float*)d_in[6];
  const float* up_as  = (const float*)d_in[7];
  const float* up_ad  = (const float*)d_in[8];
  const float* up_b   = (const float*)d_in[9];
  const float* dn_W   = (const float*)d_in[10];
  const float* dn_as  = (const float*)d_in[11];
  const float* dn_ad  = (const float*)d_in[12];
  const float* dn_b   = (const float*)d_in[13];
  const float* mlp_W  = (const float*)d_in[14];
  const float* mlp_b  = (const float*)d_in[15];
  float* out = (float*)d_out;

  char* ws = (char*)d_ws;
  float* wvec = (float*)ws;
  char* br[2] = { ws + WVEC_SZ, ws + WVEC_SZ + BR_SZ };

  k_zero<<<2048, 256, 0, stream>>>((float*)ws, WS_TOTAL / 4);
  k_prep<<<1, 64, 0, stream>>>(up_W, up_as, up_ad, dn_W, dn_as, dn_ad, wvec);

  const float* xs[2] = { up_x, dn_x };
  const int*   eis[2] = { up_ei, dn_ei };
  const float* Ws[2] = { up_W, dn_W };
  const float* bs[2] = { up_b, dn_b };

  for (int b = 0; b < 2; ++b) {
    float*    a_src  = (float*)(br[b] + BR_ASRC);
    float*    a_dst  = (float*)(br[b] + BR_ADST);
    unsigned* m_enc  = (unsigned*)(br[b] + BR_MENC);
    float*    denom  = (float*)(br[b] + BR_DEN);
    unsigned* cnt    = (unsigned*)(br[b] + BR_CNT);
    int*      csrc   = (int*)(br[b] + BR_CSRC);
    int*      cg     = (int*)(br[b] + BR_CG);
    float*    cp     = (float*)(br[b] + BR_CP);
    float*    xagg   = (float*)(br[b] + BR_XAGG);
    float*    sig    = (float*)(br[b] + BR_SIG);
    const float* wsrc = wvec + (b ? 2 : 0) * S;
    const float* wdst = wvec + (b ? 3 : 1) * S;

    k_av<<<(N * 16) / 256, 256, 0, stream>>>(xs[b], wsrc, wdst, a_src, a_dst);
    k_edge_filter<<<E / 256, 256, 0, stream>>>(eis[b], a_src, a_dst, m_enc, cnt,
                                               csrc, cg, cp);
    k_edge_soft<<<CAP / 256, 256, 0, stream>>>(cnt, cg, m_enc, cp, denom);
    k_edge_agg<<<(CAP * 16) / 256, 256, 0, stream>>>(cnt, csrc, cg, cp, denom,
                                                     xs[b], xagg);
    k_gat_out<<<G / 16, 256, 0, stream>>>(xagg, Ws[b], bs[b], sig);
  }

  float* sig_up = (float*)(br[0] + BR_SIG);
  float* sig_dn = (float*)(br[1] + BR_SIG);
  k_final<<<(G * 32) / 256, 256, 0, stream>>>(sig_up, sig_dn, mlp_W, mlp_b, out);
}
